// HeteroGNN_62981400429145
// MI455X (gfx1250) — matrix operand order repaired and checked by reading, not compiled, not run on hardware
//
#include <hip/hip_runtime.h>
#include <cstdint>
#include <cstddef>

// ---------------------------------------------------------------------------
// HeteroGNN (2-layer SAGE) for MI455X / gfx1250.
// aggregate-then-GEMM: fp32 atomic scatter (L2-resident) for mean aggregation,
// v_wmma_f32_16x16x32_f16 for all dense linears.  Weights pre-packed once per
// launch into B-fragment layout (f16, 32B/lane contiguous -> 2x b128 loads).
// Each SAGE output is produced by a FUSED dual-A GEMM:
//     C (=|+=) rowScale*A1@W1 + A2@W2 + bias   [relu]
// so each wave runs 4 WMMAs/slab over K1+K2 slabs on a 16x64 output strip,
// and the epilogue executes once per sage (not once per linear).
// ---------------------------------------------------------------------------

typedef _Float16 v16h __attribute__((ext_vector_type(16)));
typedef float    v8f  __attribute__((ext_vector_type(8)));

#define N_PAPER  50000
#define N_AUTHOR 25000
#define D_PAPER  256
#define D_AUTHOR 128
#define HID      256
#define NEDGE    400000

// ----------------------------- utility kernels -----------------------------

__global__ void zero_kernel(float4* __restrict__ p, size_t n4) {
  size_t i = (size_t)blockIdx.x * blockDim.x + threadIdx.x;
  size_t stride = (size_t)gridDim.x * blockDim.x;
  const float4 z = {0.0f, 0.0f, 0.0f, 0.0f};
  for (; i < n4; i += stride) p[i] = z;
}

__global__ void count_kernel(const int* __restrict__ dst, float* __restrict__ cnt, int E) {
  int i = blockIdx.x * blockDim.x + threadIdx.x;
  int stride = gridDim.x * blockDim.x;
  for (; i < E; i += stride)
    __hip_atomic_fetch_add(&cnt[dst[i]], 1.0f, __ATOMIC_RELAXED, __HIP_MEMORY_SCOPE_AGENT);
}

__global__ void invert_kernel(float* __restrict__ c, int n) {
  int i = blockIdx.x * blockDim.x + threadIdx.x;
  int stride = gridDim.x * blockDim.x;
  for (; i < n; i += stride) c[i] = 1.0f / fmaxf(c[i], 1.0f);
}

// Scatter-add of source rows into per-dst accumulation buffer.
// D is a compile-time constant (128 or 256) so edge/chunk decomposition is
// pure shift/mask; one thread = one (edge, float4 chunk).
template <int D>
__global__ void scatter_add_kernel(const float* __restrict__ x,
                                   const int* __restrict__ src,
                                   const int* __restrict__ dst,
                                   float* __restrict__ agg,
                                   int E) {
  constexpr int CHUNKS = D >> 2;                  // float4 chunks per row
  constexpr int SHIFT  = (D == 256) ? 6 : 5;      // log2(CHUNKS)
  const size_t total = (size_t)E << SHIFT;
  size_t t = (size_t)blockIdx.x * blockDim.x + threadIdx.x;
  const size_t stride = (size_t)gridDim.x * blockDim.x;
  for (; t < total; t += stride) {
    const size_t e = t >> SHIFT;
    const int c = (int)(t & (CHUNKS - 1)) << 2;
    const int s = src[e];
    const int d = dst[e];
    const float4 v = *reinterpret_cast<const float4*>(x + (size_t)s * D + c);
    float* p = agg + (size_t)d * D + c;
    __hip_atomic_fetch_add(p + 0, v.x, __ATOMIC_RELAXED, __HIP_MEMORY_SCOPE_AGENT);
    __hip_atomic_fetch_add(p + 1, v.y, __ATOMIC_RELAXED, __HIP_MEMORY_SCOPE_AGENT);
    __hip_atomic_fetch_add(p + 2, v.z, __ATOMIC_RELAXED, __HIP_MEMORY_SCOPE_AGENT);
    __hip_atomic_fetch_add(p + 3, v.w, __ATOMIC_RELAXED, __HIP_MEMORY_SCOPE_AGENT);
  }
}

// ------------------------- weight fragment packing --------------------------
__global__ void pack_w_kernel(const float* __restrict__ W, _Float16* __restrict__ PW,
                              int K, int N) {
  const int KS = K >> 5;
  const int NT = N >> 4;
  const int total = NT * KS * 32;
  int t = blockIdx.x * blockDim.x + threadIdx.x;
  int stride = gridDim.x * blockDim.x;
  for (; t < total; t += stride) {
    const int lane = t & 31;
    const int rest = t >> 5;
    const int ks = rest % KS;
    const int nt = rest / KS;
    const int l16 = lane & 15;
    const int bkb = (lane >> 4) << 3;            // 0 or 8
    const float* src = W + (size_t)(ks * 32 + bkb) * N + nt * 16 + l16;
    _Float16* dstp = PW + ((size_t)(nt * KS + ks) * 32 + lane) * 16;
#pragma unroll
    for (int h = 0; h < 16; ++h) dstp[h] = (_Float16)src[(size_t)(h + (h & 8)) * N];
  }
}

// ------------------------------- WMMA GEMM ---------------------------------
// Accumulate KS slabs of  scale*Arow @ PW  into c0..c3 (4 n-tiles).
__device__ __forceinline__ void mm_slabs(const float* __restrict__ arow, float scale,
                                         const _Float16* __restrict__ PW, int KS,
                                         int nt0, int lane, int akb,
                                         v8f& c0, v8f& c1, v8f& c2, v8f& c3) {
  for (int ks = 0; ks < KS; ++ks) {
    v16h a;
    const float* ap = arow + ks * 32 + akb;
#pragma unroll
    for (int h = 0; h < 8; ++h) a[h]     = (_Float16)(ap[h] * scale);
#pragma unroll
    for (int h = 0; h < 8; ++h) a[8 + h] = (_Float16)(ap[16 + h] * scale);

    const size_t fb = ((size_t)nt0 * KS + ks) * 32 + lane;
    const v16h b0 = *(const v16h*)(PW + (fb)           * 16);
    const v16h b1 = *(const v16h*)(PW + (fb + KS * 32) * 16);
    const v16h b2 = *(const v16h*)(PW + (fb + KS * 64) * 16);
    const v16h b3 = *(const v16h*)(PW + (fb + KS * 96) * 16);

    c0 = __builtin_amdgcn_wmma_f32_16x16x32_f16(false, a, false, b0, (short)0, c0, false, false);
    c1 = __builtin_amdgcn_wmma_f32_16x16x32_f16(false, a, false, b1, (short)0, c1, false, false);
    c2 = __builtin_amdgcn_wmma_f32_16x16x32_f16(false, a, false, b2, (short)0, c2, false, false);
    c3 = __builtin_amdgcn_wmma_f32_16x16x32_f16(false, a, false, b3, (short)0, c3, false, false);
  }
}

// C[M,N] (= or +=) rowScale[m]*A1[M,K1]@W1 + A2[M,K2]@W2 + bias[n]   [relu]
// One wave32 -> 16x64 strip (4 x 16x16 tiles).
template <int ACC, int RELU>
__global__ __launch_bounds__(128)
void gemm2_wmma_kernel(const float* __restrict__ A1,
                       const float* __restrict__ rowScale,
                       const _Float16* __restrict__ PW1, int K1,
                       const float* __restrict__ A2,
                       const _Float16* __restrict__ PW2, int K2,
                       const float* __restrict__ bias,
                       float* __restrict__ C,
                       int M, int N) {
  const int wave = threadIdx.x >> 5;
  const int lane = threadIdx.x & 31;
  const int hi   = lane >> 4;
  const int l16  = lane & 15;

  const int m0  = (blockIdx.x * 4 + wave) * 16;
  const int nt0 = blockIdx.y * 4;
  if (m0 >= M) return;                            // wave-uniform

  // Rows >= M are never stored, so just clamp the row index for loads.
  int aRow = m0 + l16;
  aRow = (aRow < M) ? aRow : (M - 1);
  const float rs = rowScale[aRow];
  const float* arow1 = A1 + (size_t)aRow * K1;
  const float* arow2 = A2 + (size_t)aRow * K2;
  const int akb = hi << 3;                        // 0 or 8

  v8f c0 = {}, c1 = {}, c2 = {}, c3 = {};
  mm_slabs(arow1, rs,   PW1, K1 >> 5, nt0, lane, akb, c0, c1, c2, c3);
  mm_slabs(arow2, 1.0f, PW2, K2 >> 5, nt0, lane, akb, c0, c1, c2, c3);

  v8f acc[4] = {c0, c1, c2, c3};
  const int mbase = m0 + hi * 8;

  if (m0 + 16 <= M) {
    // full tile: no per-element guards
#pragma unroll
    for (int j = 0; j < 4; ++j) {
      const int bn = (nt0 + j) * 16 + l16;
      const float bv = bias[bn];
      float* cp = C + (size_t)mbase * N + bn;
#pragma unroll
      for (int r = 0; r < 8; ++r) {
        float v = acc[j][r] + bv;
        if (ACC)  v += cp[(size_t)r * N];
        if (RELU) v = fmaxf(v, 0.0f);
        cp[(size_t)r * N] = v;
      }
    }
  } else {
#pragma unroll
    for (int j = 0; j < 4; ++j) {
      const int bn = (nt0 + j) * 16 + l16;
      const float bv = bias[bn];
#pragma unroll
      for (int r = 0; r < 8; ++r) {
        const int m = mbase + r;
        if (m < M) {
          const size_t off = (size_t)m * N + bn;
          float v = acc[j][r] + bv;
          if (ACC)  v += C[off];
          if (RELU) v = fmaxf(v, 0.0f);
          C[off] = v;
        }
      }
    }
  }
}

// ------------------------------- host driver -------------------------------

extern "C" void kernel_launch(void* const* d_in, const int* in_sizes, int n_in,
                              void* d_out, int out_size, void* d_ws, size_t ws_size,
                              hipStream_t stream) {
  const float* x_paper   = (const float*)d_in[0];
  const float* x_author  = (const float*)d_in[1];
  const int* cites_src   = (const int*)d_in[2];
  const int* cites_dst   = (const int*)d_in[3];
  const int* writes_src  = (const int*)d_in[4];
  const int* writes_dst  = (const int*)d_in[5];
  const int* rev_src     = (const int*)d_in[6];
  const int* rev_dst     = (const int*)d_in[7];
  const float* Wf32[12] = {
    (const float*)d_in[8],  (const float*)d_in[10],   // l1c_Wl, l1c_Wr
    (const float*)d_in[11], (const float*)d_in[13],   // l1w_Wl, l1w_Wr
    (const float*)d_in[14], (const float*)d_in[16],   // l1r_Wl, l1r_Wr
    (const float*)d_in[17], (const float*)d_in[19],   // l2c_Wl, l2c_Wr
    (const float*)d_in[20], (const float*)d_in[22],   // l2w_Wl, l2w_Wr
    (const float*)d_in[23], (const float*)d_in[25],   // l2r_Wl, l2r_Wr
  };
  const int WK[12] = {256, 256, 128, 256, 256, 128, 256, 256, 256, 256, 256, 256};
  const float* l1c_bl = (const float*)d_in[9];
  const float* l1w_bl = (const float*)d_in[12];
  const float* l1r_bl = (const float*)d_in[15];
  const float* l2c_bl = (const float*)d_in[18];
  const float* l2w_bl = (const float*)d_in[21];
  const float* l2r_bl = (const float*)d_in[24];

  // workspace layout
  float* agg  = (float*)d_ws;                                 // 50000*256 f32
  float* p1   = agg  + (size_t)N_PAPER  * HID;                // 50000*256 f32
  float* a1   = p1   + (size_t)N_PAPER  * HID;                // 25000*256 f32
  float* invc = a1   + (size_t)N_AUTHOR * HID;                // 50000 f32
  float* invw = invc + N_PAPER;                               // 50000 f32
  float* invr = invw + N_PAPER;                               // 25000 f32
  _Float16* pwbase = (_Float16*)(invr + N_AUTHOR);
  _Float16* PW[12];
  {
    _Float16* p = pwbase;
    for (int i = 0; i < 12; ++i) { PW[i] = p; p += (size_t)WK[i] * HID; }
  }

  float* p2 = (float*)d_out;                                  // 50000*256
  float* a2 = p2 + (size_t)N_PAPER * HID;                     // 25000*256

  auto zero = [&](float* p, size_t n) {
    zero_kernel<<<2048, 256, 0, stream>>>((float4*)p, n >> 2);
  };
  auto scatter256 = [&](const float* x, const int* src, const int* dst, float* out) {
    scatter_add_kernel<256><<<8192, 256, 0, stream>>>(x, src, dst, out, NEDGE);
  };
  auto scatter128 = [&](const float* x, const int* src, const int* dst, float* out) {
    scatter_add_kernel<128><<<8192, 256, 0, stream>>>(x, src, dst, out, NEDGE);
  };
  // fused sage GEMM: C (=|+=) rs*A1@W1 + A2@W2 + b   [relu]
  auto sage = [&](const float* A1, const float* rsc, const _Float16* pw1, int K1,
                  const float* A2, const _Float16* pw2, int K2,
                  const float* b, float* Cout, int M, int acc, int relu) {
    dim3 grid((unsigned)((M + 63) / 64), (unsigned)(HID / 64));
    if (!acc && !relu)
      gemm2_wmma_kernel<0, 0><<<grid, 128, 0, stream>>>(A1, rsc, pw1, K1, A2, pw2, K2, b, Cout, M, HID);
    else if (!acc && relu)
      gemm2_wmma_kernel<0, 1><<<grid, 128, 0, stream>>>(A1, rsc, pw1, K1, A2, pw2, K2, b, Cout, M, HID);
    else if (acc && !relu)
      gemm2_wmma_kernel<1, 0><<<grid, 128, 0, stream>>>(A1, rsc, pw1, K1, A2, pw2, K2, b, Cout, M, HID);
    else
      gemm2_wmma_kernel<1, 1><<<grid, 128, 0, stream>>>(A1, rsc, pw1, K1, A2, pw2, K2, b, Cout, M, HID);
  };

  // ---- pack all 12 weight matrices into B-fragment layout ----
  for (int i = 0; i < 12; ++i) {
    int total = (HID / 16) * (WK[i] / 32) * 32;
    pack_w_kernel<<<(total + 255) / 256, 256, 0, stream>>>(Wf32[i], PW[i], WK[i], HID);
  }

  // ---- in-degree reciprocals (shared by both layers) ----
  zero(invc, N_PAPER);  zero(invw, N_PAPER);  zero(invr, N_AUTHOR);
  count_kernel<<<1024, 256, 0, stream>>>(cites_dst,  invc, NEDGE);
  count_kernel<<<1024, 256, 0, stream>>>(writes_dst, invw, NEDGE);
  count_kernel<<<1024, 256, 0, stream>>>(rev_dst,    invr, NEDGE);
  invert_kernel<<<256, 256, 0, stream>>>(invc, N_PAPER);
  invert_kernel<<<256, 256, 0, stream>>>(invw, N_PAPER);
  invert_kernel<<<256, 256, 0, stream>>>(invr, N_AUTHOR);

  // ---- layer 1, paper dst: p1 = relu(sage_cites + sage_writes) ----
  zero(agg, (size_t)N_PAPER * D_PAPER);
  scatter256(x_paper, cites_src, cites_dst, agg);
  sage(agg, invc, PW[0], D_PAPER, x_paper, PW[1], D_PAPER, l1c_bl, p1, N_PAPER, 0, 0);
  zero(agg, (size_t)N_PAPER * D_AUTHOR);
  scatter128(x_author, writes_src, writes_dst, agg);
  sage(agg, invw, PW[2], D_AUTHOR, x_paper, PW[3], D_PAPER, l1w_bl, p1, N_PAPER, 1, 1);

  // ---- layer 1, author dst: a1 = relu(sage_rev) ----
  zero(agg, (size_t)N_AUTHOR * D_PAPER);
  scatter256(x_paper, rev_src, rev_dst, agg);
  sage(agg, invr, PW[4], D_PAPER, x_author, PW[5], D_AUTHOR, l1r_bl, a1, N_AUTHOR, 0, 1);

  // ---- layer 2, paper dst -> p2 (d_out) ----
  zero(agg, (size_t)N_PAPER * HID);
  scatter256(p1, cites_src, cites_dst, agg);
  sage(agg, invc, PW[6], HID, p1, PW[7], HID, l2c_bl, p2, N_PAPER, 0, 0);
  zero(agg, (size_t)N_PAPER * HID);
  scatter256(a1, writes_src, writes_dst, agg);
  sage(agg, invw, PW[8], HID, p1, PW[9], HID, l2w_bl, p2, N_PAPER, 1, 0);

  // ---- layer 2, author dst -> a2 (d_out) ----
  zero(agg, (size_t)N_AUTHOR * HID);
  scatter256(p1, rev_src, rev_dst, agg);
  sage(agg, invr, PW[10], HID, a1, PW[11], HID, l2r_bl, a2, N_AUTHOR, 0, 0);
}
